// HeteroGNN_TransformerConv_26293789786130
// MI455X (gfx1250) — compile-verified, matrix operand order checked by reading, not run on hardware
//
#include <hip/hip_runtime.h>
#include <hip/hip_bf16.h>

// ---------------------------------------------------------------------------
// Problem constants (match reference)
// ---------------------------------------------------------------------------
#define N_GUIDE 20000
#define N_SITE  200000
#define NEDGE   400000
#define E_LAB   200000
#define GIN     128
#define HIDN    64
#define EDIM    25
#define NL      3

typedef __attribute__((ext_vector_type(16))) _Float16 v16h;
typedef __attribute__((ext_vector_type(8)))  _Float16 v8h;
typedef __attribute__((ext_vector_type(2)))  _Float16 v2h;
typedef __attribute__((ext_vector_type(8)))  float    v8f;

// ---------------------------------------------------------------------------
// Param offsets into the flattened params pytree (JAX flatten: sorted keys,
// recursively; all leaves f32, concatenated).  Units: floats.
// ---------------------------------------------------------------------------
static constexpr size_t P_CONV_R = 0;
static constexpr size_t P_CONV_T = 62400;
// within a conv block (keys sorted: We,Wk,Wq,Ws,Wv,be,bk,bq,bs,bv), W:(3,64,64), b:(3,64)
static constexpr size_t CV_We = 0, CV_Wk = 12288, CV_Wq = 24576, CV_Ws = 36864, CV_Wv = 49152;
static constexpr size_t CV_be = 61440, CV_bk = 61632, CV_bq = 61824, CV_bs = 62016, CV_bv = 62208;
static constexpr size_t P_DEC_W1 = 124800, P_DEC_W2 = 149376, P_DEC_W3 = 157568;
static constexpr size_t P_DEC_B1 = 157632, P_DEC_B2 = 157760, P_DEC_B3 = 157824;
static constexpr size_t P_DEC_LN1_B = 157825, P_DEC_LN1_G = 157953;
static constexpr size_t P_DEC_LN2_B = 158081, P_DEC_LN2_G = 158145;
static constexpr size_t P_EF_W1 = 158209, P_EF_W2 = 159809, P_EF_B1 = 163905, P_EF_B2 = 163969;
static constexpr size_t P_EF_BN1_B = 164033, P_EF_BN1_G = 164097, P_EF_BN1_M = 164161, P_EF_BN1_V = 164225;
static constexpr size_t P_EF_BN2_B = 164289, P_EF_BN2_G = 164353, P_EF_BN2_M = 164417, P_EF_BN2_V = 164481;
static constexpr size_t P_ENC_G_W = 164545, P_ENC_G_B = 172737;
static constexpr size_t P_ENC_S_W = 172801, P_ENC_S_B = 180993;
static constexpr size_t P_LNL_G_B = 181057, P_LNL_G_G = 181249;
static constexpr size_t P_LNL_S_B = 181441, P_LNL_S_G = 181633;
static constexpr size_t P_LN_G_B = 181825, P_LN_G_G = 181889;
static constexpr size_t P_LN_S_B = 181953, P_LN_S_G = 182017;
static constexpr size_t P_MP_W = 182081, P_MP_B = 183681;

__device__ __forceinline__ v8h cvt8(float4 u0, float4 u1)
{
    v8h hv;
    hv[0] = (_Float16)u0.x; hv[1] = (_Float16)u0.y;
    hv[2] = (_Float16)u0.z; hv[3] = (_Float16)u0.w;
    hv[4] = (_Float16)u1.x; hv[5] = (_Float16)u1.y;
    hv[6] = (_Float16)u1.z; hv[7] = (_Float16)u1.w;
    return hv;
}

// ---------------------------------------------------------------------------
// Generic WMMA GEMM:  C[M,N] = act( A[M,K] @ W[K,N] + bias )
//   act: 0 = none, 1 = relu, 2 = BN(eval) + relu
//   KVEC (compile-time): K % 32 == 0 and rows 16-B aligned -> fully
//   branch-free vector staging for full tiles (block-uniform rowFull branch).
//   block = 256 thr (8 waves); tile = 128 rows x 64 cols; K staged by 32.
//   LDS row stride = 40 halves (80B): conflict-free b128 fragment loads.
// ---------------------------------------------------------------------------
template <bool KVEC>
__global__ __launch_bounds__(256)
void gemm_wmma(const float* __restrict__ A, const float* __restrict__ W,
               const float* __restrict__ bias, float* __restrict__ C,
               int M, int N, int K, int act,
               const float* __restrict__ bn_g, const float* __restrict__ bn_b,
               const float* __restrict__ bn_m, const float* __restrict__ bn_v)
{
    __shared__ _Float16 ldsA[128 * 40];
    __shared__ _Float16 ldsB[64 * 40];

    const int tid  = threadIdx.x;
    const int wv   = tid >> 5;
    const int lane = tid & 31;
    const int rowBase = blockIdx.x * 128;
    const int colBase = blockIdx.y * 64;
    const bool rowFull = (rowBase + 128 <= M);   // block-uniform

    const int mloc = wv * 16 + (lane & 15);      // A fragment row within tile
    const int kb0  = (lane >> 4) << 3;           // per-lane K sub-base {0,8}

    // A staging coordinates: 512 chunks of 8 halves; this thread owns 2.
    const int ar0 = tid >> 2;                    // rows 0..63   (chunk set 0)
    const int ar1 = ar0 + 64;                    // rows 64..127 (chunk set 1)
    const int akc = (tid & 3) << 3;              // k-offset within tile: 0/8/16/24
    // B staging coordinates: (2k x 4n) micro-tile per thread.
    const int bkp = tid >> 4;                    // k-pair 0..15
    const int bn4 = (tid & 15) << 2;             // n 0..60 step 4

    v8f acc[4];
    #pragma unroll
    for (int i = 0; i < 4; ++i) acc[i] = v8f{0.f,0.f,0.f,0.f,0.f,0.f,0.f,0.f};

    for (int kb = 0; kb < K; kb += 32) {
        if (KVEC && rowFull) {
            // ---- branch-free vector staging (hot path) ----
            #pragma unroll
            for (int c = 0; c < 2; ++c) {
                int r = (c == 0) ? ar0 : ar1;
                const float* ap = A + (size_t)(rowBase + r) * K + kb + akc;
                const float4* p = (const float4*)__builtin_assume_aligned(ap, 16);
                *(v8h*)&ldsA[r * 40 + akc] = cvt8(p[0], p[1]);
            }
            const float* wp = W + (size_t)(kb + 2 * bkp) * N + colBase + bn4;
            float4 a0 = *(const float4*)__builtin_assume_aligned(wp, 16);
            float4 a1 = *(const float4*)__builtin_assume_aligned(wp + N, 16);
            v2h p0 = { (_Float16)a0.x, (_Float16)a1.x };
            v2h p1 = { (_Float16)a0.y, (_Float16)a1.y };
            v2h p2 = { (_Float16)a0.z, (_Float16)a1.z };
            v2h p3 = { (_Float16)a0.w, (_Float16)a1.w };
            *(v2h*)&ldsB[(bn4 + 0) * 40 + 2 * bkp] = p0;
            *(v2h*)&ldsB[(bn4 + 1) * 40 + 2 * bkp] = p1;
            *(v2h*)&ldsB[(bn4 + 2) * 40 + 2 * bkp] = p2;
            *(v2h*)&ldsB[(bn4 + 3) * 40 + 2 * bkp] = p3;
        } else {
            // ---- guarded scalar staging (edge tiles / K=25) ----
            #pragma unroll
            for (int c = 0; c < 2; ++c) {
                int r  = (c == 0) ? ar0 : ar1;
                int gr = rowBase + r;
                int gk = kb + akc;
                v8h hv;
                #pragma unroll
                for (int j = 0; j < 8; ++j) {
                    int gkj = gk + j;
                    float v = (gr < M && gkj < K) ? A[(size_t)gr * K + gkj] : 0.f;
                    hv[j] = (_Float16)v;
                }
                *(v8h*)&ldsA[r * 40 + akc] = hv;
            }
            int gk0 = kb + 2 * bkp, gk1 = gk0 + 1;
            float4 z4; z4.x = z4.y = z4.z = z4.w = 0.f;
            float4 a0 = (gk0 < K) ? *(const float4*)(W + (size_t)gk0 * N + colBase + bn4) : z4;
            float4 a1 = (gk1 < K) ? *(const float4*)(W + (size_t)gk1 * N + colBase + bn4) : z4;
            v2h p0 = { (_Float16)a0.x, (_Float16)a1.x };
            v2h p1 = { (_Float16)a0.y, (_Float16)a1.y };
            v2h p2 = { (_Float16)a0.z, (_Float16)a1.z };
            v2h p3 = { (_Float16)a0.w, (_Float16)a1.w };
            *(v2h*)&ldsB[(bn4 + 0) * 40 + 2 * bkp] = p0;
            *(v2h*)&ldsB[(bn4 + 1) * 40 + 2 * bkp] = p1;
            *(v2h*)&ldsB[(bn4 + 2) * 40 + 2 * bkp] = p2;
            *(v2h*)&ldsB[(bn4 + 3) * 40 + 2 * bkp] = p3;
        }
        __syncthreads();

        // A fragment: elems 0-7 = K[kb0..kb0+7], elems 8-15 = K[kb0+16..+23]
        v8h alo = *(const v8h*)&ldsA[mloc * 40 + kb0];
        v8h ahi = *(const v8h*)&ldsA[mloc * 40 + kb0 + 16];
        v16h afrag = __builtin_shufflevector(alo, ahi,
            0,1,2,3,4,5,6,7,8,9,10,11,12,13,14,15);

        // Preload all four B fragments, then issue the WMMAs back-to-back.
        v16h bfrag[4];
        #pragma unroll
        for (int nt = 0; nt < 4; ++nt) {
            int nloc = nt * 16 + (lane & 15);
            v8h blo = *(const v8h*)&ldsB[nloc * 40 + kb0];
            v8h bhi = *(const v8h*)&ldsB[nloc * 40 + kb0 + 16];
            bfrag[nt] = __builtin_shufflevector(blo, bhi,
                0,1,2,3,4,5,6,7,8,9,10,11,12,13,14,15);
        }
        #pragma unroll
        for (int nt = 0; nt < 4; ++nt)
            acc[nt] = __builtin_amdgcn_wmma_f32_16x16x32_f16(
                false, afrag, false, bfrag[nt], (short)0, acc[nt], false, false);
        __syncthreads();
    }

    // ---- epilogue: bias (+BN) (+relu), store f32 ----
    #pragma unroll
    for (int nt = 0; nt < 4; ++nt) {
        int ncol = colBase + nt * 16 + (lane & 15);
        float bv_ = bias ? bias[ncol] : 0.f;
        float sc = 1.f, sh = 0.f;
        if (act == 2) {
            float s_ = bn_g[ncol] * rsqrtf(bn_v[ncol] + 1e-5f);
            sc = s_;
            sh = bn_b[ncol] - bn_m[ncol] * s_;
        }
        #pragma unroll
        for (int r = 0; r < 8; ++r) {
            int row = rowBase + wv * 16 + ((lane >> 4) << 3) + r;
            if (row < M) {
                float v = acc[nt][r] + bv_;
                if (act == 2) v = v * sc + sh;
                if (act >= 1) v = fmaxf(v, 0.f);
                C[(size_t)row * N + ncol] = v;
            }
        }
    }
}

// ---------------------------------------------------------------------------
// Row LayerNorm, wave32 per row, WIDTH/32 elems per lane (in-place safe).
// ---------------------------------------------------------------------------
template <int WIDTH>
__global__ __launch_bounds__(256)
void ln_kernel(const float* __restrict__ x, float* __restrict__ out,
               const float* __restrict__ g, const float* __restrict__ b, int M)
{
    constexpr int P = WIDTH / 32;
    int wv = threadIdx.x >> 5, lane = threadIdx.x & 31;
    int row = blockIdx.x * 8 + wv;
    if (row >= M) return;
    const float* xr = x + (size_t)row * WIDTH;
    float v[P];
    float s = 0.f, s2 = 0.f;
    #pragma unroll
    for (int p = 0; p < P; ++p) {
        v[p] = xr[lane + p * 32];
        s += v[p]; s2 += v[p] * v[p];
    }
    #pragma unroll
    for (int o = 16; o; o >>= 1) { s += __shfl_xor(s, o, 32); s2 += __shfl_xor(s2, o, 32); }
    float mu = s / WIDTH;
    float rinv = rsqrtf(s2 / WIDTH - mu * mu + 1e-5f);
    #pragma unroll
    for (int p = 0; p < P; ++p) {
        int c = lane + p * 32;
        out[(size_t)row * WIDTH + c] = (v[p] - mu) * rinv * g[c] + b[c];
    }
}

// ---------------------------------------------------------------------------
// Order-preserving float <-> uint key (for atomicMax-based segment max)
// ---------------------------------------------------------------------------
__device__ __forceinline__ unsigned fkey(float f) {
    unsigned b = __float_as_uint(f);
    return (b & 0x80000000u) ? ~b : (b | 0x80000000u);
}
__device__ __forceinline__ float fdec(unsigned k) {
    return __uint_as_float((k & 0x80000000u) ? (k ^ 0x80000000u) : ~k);
}

// ---------------------------------------------------------------------------
// Attention pass 1: per edge, 4 head scores; segment max via atomicMax(key)
// ---------------------------------------------------------------------------
__global__ __launch_bounds__(256)
void attn_pass1(const float* __restrict__ q, const float* __restrict__ k,
                const float* __restrict__ e, const int* __restrict__ src,
                const int* __restrict__ dst, float* __restrict__ score,
                unsigned* __restrict__ mkey, int nE)
{
    int eid = blockIdx.x * 256 + threadIdx.x;
    if (eid >= nE) return;
    int s = src[eid], d = dst[eid];
    const float4* qp = (const float4*)(q + (size_t)d * 64);
    const float4* kp = (const float4*)(k + (size_t)s * 64);
    const float4* ep = (const float4*)(e + (size_t)eid * 64);
    float acc[4] = {0.f, 0.f, 0.f, 0.f};
    #pragma unroll
    for (int i = 0; i < 16; ++i) {
        float4 qv = qp[i], kv = kp[i], ev = ep[i];
        acc[i >> 2] += qv.x * (kv.x + ev.x) + qv.y * (kv.y + ev.y)
                     + qv.z * (kv.z + ev.z) + qv.w * (kv.w + ev.w);
    }
    #pragma unroll
    for (int h = 0; h < 4; ++h) {
        float sc = acc[h] * 0.25f;                 // 1/sqrt(16)
        score[(size_t)eid * 4 + h] = sc;
        atomicMax(&mkey[(size_t)d * 4 + h], fkey(sc));
    }
}

// ---------------------------------------------------------------------------
// Attention pass 2: w = exp(s - max); z[dst] += w; agg[dst] += w*(v[src]+e)
// ---------------------------------------------------------------------------
__global__ __launch_bounds__(256)
void attn_pass2(const float* __restrict__ v, const float* __restrict__ e,
                const int* __restrict__ src, const int* __restrict__ dst,
                const float* __restrict__ score, const unsigned* __restrict__ mkey,
                float* __restrict__ z, float* __restrict__ agg, int nE)
{
    int eid = blockIdx.x * 256 + threadIdx.x;
    if (eid >= nE) return;
    int s = src[eid], d = dst[eid];
    float w[4];
    #pragma unroll
    for (int h = 0; h < 4; ++h) {
        float m = fdec(mkey[(size_t)d * 4 + h]);
        w[h] = __expf(score[(size_t)eid * 4 + h] - m);
        atomicAdd(&z[(size_t)d * 4 + h], w[h]);
    }
    const float4* vp = (const float4*)(v + (size_t)s * 64);
    const float4* ep = (const float4*)(e + (size_t)eid * 64);
    #pragma unroll
    for (int i = 0; i < 16; ++i) {
        float4 vv = vp[i], ev = ep[i];
        float ww = w[i >> 2];
        float* ap = agg + (size_t)d * 64 + i * 4;
        atomicAdd(ap + 0, ww * (vv.x + ev.x));
        atomicAdd(ap + 1, ww * (vv.y + ev.y));
        atomicAdd(ap + 2, ww * (vv.z + ev.z));
        atomicAdd(ap + 3, ww * (vv.w + ev.w));
    }
}

// ---------------------------------------------------------------------------
// Node update:  h = LN( relu(agg/z + skip) + h ), width 64, wave32 per row
// ---------------------------------------------------------------------------
__global__ __launch_bounds__(256)
void node_update(const float* __restrict__ agg, const float* __restrict__ z,
                 const float* __restrict__ skip, const float* __restrict__ g,
                 const float* __restrict__ b, float* __restrict__ h, int M)
{
    int wv = threadIdx.x >> 5, lane = threadIdx.x & 31;
    int row = blockIdx.x * 8 + wv;
    if (row >= M) return;
    size_t base = (size_t)row * 64;
    float r[2];
    #pragma unroll
    for (int p = 0; p < 2; ++p) {
        int d = lane + p * 32;
        float zz = z[(size_t)row * 4 + (d >> 4)];
        float a = (zz > 0.f) ? agg[base + d] / zz : 0.f;
        float val = a + skip[base + d];
        r[p] = fmaxf(val, 0.f) + h[base + d];
    }
    float s = r[0] + r[1], s2 = r[0] * r[0] + r[1] * r[1];
    #pragma unroll
    for (int o = 16; o; o >>= 1) { s += __shfl_xor(s, o, 32); s2 += __shfl_xor(s2, o, 32); }
    float mu = s * (1.f / 64.f);
    float rinv = rsqrtf(s2 * (1.f / 64.f) - mu * mu + 1e-5f);
    #pragma unroll
    for (int p = 0; p < 2; ++p) {
        int d = lane + p * 32;
        h[base + d] = (r[p] - mu) * rinv * g[d] + b[d];
    }
}

// ---------------------------------------------------------------------------
// Decoder gather/concat: zc[i] = [hg[gi[i]] | hs[si[i]] | h2[i]]  (192 wide)
// ---------------------------------------------------------------------------
__global__ __launch_bounds__(256)
void gather_concat(const float* __restrict__ hg, const float* __restrict__ hs,
                   const float* __restrict__ h2, const int* __restrict__ gi,
                   const int* __restrict__ si, float* __restrict__ zc, int M)
{
    int wv = threadIdx.x >> 5, lane = threadIdx.x & 31;
    int i = blockIdx.x * 8 + wv;
    if (i >= M) return;
    int g = gi[i], s = si[i];
    float* o = zc + (size_t)i * 192;
    #pragma unroll
    for (int j = 0; j < 2; ++j) {
        int c = j * 32 + lane;
        o[c]        = hg[(size_t)g * 64 + c];
        o[64 + c]   = hs[(size_t)s * 64 + c];
        o[128 + c]  = h2[(size_t)i * 64 + c];
    }
}

// ---------------------------------------------------------------------------
// Final: out[i] = dot(d2[i], w3) + b3  (wave32 per row)
// ---------------------------------------------------------------------------
__global__ __launch_bounds__(256)
void final_dot(const float* __restrict__ d2, const float* __restrict__ w3,
               const float* __restrict__ b3, float* __restrict__ out, int M)
{
    int wv = threadIdx.x >> 5, lane = threadIdx.x & 31;
    int row = blockIdx.x * 8 + wv;
    if (row >= M) return;
    size_t base = (size_t)row * 64;
    float p = d2[base + lane] * w3[lane] + d2[base + 32 + lane] * w3[32 + lane];
    #pragma unroll
    for (int o = 16; o; o >>= 1) p += __shfl_xor(p, o, 32);
    if (lane == 0) out[row] = p + b3[0];
}

// ---------------------------------------------------------------------------
// Fills
// ---------------------------------------------------------------------------
__global__ __launch_bounds__(256)
void fill_f32(float* p, float v, long n)
{
    long i = (long)blockIdx.x * 256 + threadIdx.x;
    if (i < n) p[i] = v;
}
__global__ __launch_bounds__(256)
void fill_u32(unsigned* p, unsigned v, long n)
{
    long i = (long)blockIdx.x * 256 + threadIdx.x;
    if (i < n) p[i] = v;
}

// ---------------------------------------------------------------------------
// Host orchestration
// ---------------------------------------------------------------------------
static inline void g_gemm(hipStream_t st, const float* A, const float* W,
                          const float* bias, float* C, int M, int N, int K,
                          int act, const float* g = nullptr, const float* b = nullptr,
                          const float* m = nullptr, const float* v = nullptr)
{
    dim3 grid((M + 127) / 128, N / 64);
    if ((K & 31) == 0)
        gemm_wmma<true><<<grid, 256, 0, st>>>(A, W, bias, C, M, N, K, act, g, b, m, v);
    else
        gemm_wmma<false><<<grid, 256, 0, st>>>(A, W, bias, C, M, N, K, act, g, b, m, v);
}
static inline void g_fillf(hipStream_t st, float* p, float v, long n)
{
    fill_f32<<<(int)((n + 255) / 256), 256, 0, st>>>(p, v, n);
}
static inline void g_fillu(hipStream_t st, unsigned* p, unsigned v, long n)
{
    fill_u32<<<(int)((n + 255) / 256), 256, 0, st>>>(p, v, n);
}

extern "C" void kernel_launch(void* const* d_in, const int* in_sizes, int n_in,
                              void* d_out, int out_size, void* d_ws, size_t ws_size,
                              hipStream_t stream)
{
    const float* x_guide = (const float*)d_in[0];
    const float* x_site  = (const float*)d_in[1];
    const float* eat_raw = (const float*)d_in[2];
    const float* ear_raw = (const float*)d_in[3];
    const float* efeat   = (const float*)d_in[4];
    const float* P       = (const float*)d_in[5];
    const int*   ei_src  = (const int*)d_in[6];
    const int*   ei_dst  = (const int*)d_in[7];
    const int*   lgi     = (const int*)d_in[8];
    const int*   lsi     = (const int*)d_in[9];
    float*       out     = (float*)d_out;
    (void)in_sizes; (void)n_in; (void)out_size; (void)ws_size;

    // ---- workspace arena (floats) ----
    float* ws = (float*)d_ws;
    size_t o = 0;
    auto alloc = [&](size_t n) { float* p = ws + o; o += n; return p; };
    float*    hs   = alloc((size_t)N_SITE * 64);
    float*    hg   = alloc((size_t)N_GUIDE * 64);
    float*    aggS = alloc((size_t)N_SITE * 64);    // conv_t aggregation / dec h1
    float*    aggG = alloc((size_t)N_GUIDE * 64);   // conv_r aggregation / enc tmp
    float*    eaT  = alloc((size_t)NEDGE * 64);     // relu(mp(edge_attr_targets))
    float*    eaR  = alloc((size_t)NEDGE * 64);
    float*    eP   = alloc((size_t)NEDGE * 64);     // per-layer edge projection
    float*    qS   = alloc((size_t)N_SITE * 64);
    float*    skS  = alloc((size_t)N_SITE * 64);
    float*    kS   = alloc((size_t)N_SITE * 64);
    float*    vS   = alloc((size_t)N_SITE * 64);
    float*    qG   = alloc((size_t)N_GUIDE * 64);
    float*    skG  = alloc((size_t)N_GUIDE * 64);
    float*    kG   = alloc((size_t)N_GUIDE * 64);
    float*    vG   = alloc((size_t)N_GUIDE * 64);
    float*    scr  = alloc((size_t)NEDGE * 4);      // per-edge head scores
    unsigned* mS   = (unsigned*)alloc((size_t)N_SITE * 4);
    float*    zS   = alloc((size_t)N_SITE * 4);
    unsigned* mG   = (unsigned*)alloc((size_t)N_GUIDE * 4);
    float*    zG   = alloc((size_t)N_GUIDE * 4);
    // decoder aliases (onto buffers dead after the GNN phase)
    float* h1   = aggS;            // [E_LAB,64]
    float* h2   = kS;              // [E_LAB,64]
    float* zcat = eaT;             // [E_LAB,192]  (spans eaT+part of eaR)
    float* dd1  = qS;              // [E_LAB,128]  (spans qS+skS)
    float* dd2  = vS;              // [E_LAB,64]

    const int ln_blocks_S = (N_SITE + 7) / 8;
    const int ln_blocks_G = (N_GUIDE + 7) / 8;
    const int e_blocks    = (NEDGE + 255) / 256;

    // ---- encoders: h = LN(relu(x @ W + b)) ----
    g_gemm(stream, x_guide, P + P_ENC_G_W, P + P_ENC_G_B, aggG, N_GUIDE, 64, GIN, 1);
    ln_kernel<64><<<ln_blocks_G, 256, 0, stream>>>(aggG, hg, P + P_LN_G_G, P + P_LN_G_B, N_GUIDE);
    g_gemm(stream, x_site, P + P_ENC_S_W, P + P_ENC_S_B, aggS, N_SITE, 64, GIN, 1);
    ln_kernel<64><<<ln_blocks_S, 256, 0, stream>>>(aggS, hs, P + P_LN_S_G, P + P_LN_S_B, N_SITE);

    // ---- edge MLP: ea = relu(edge_attr @ mp_W + mp_b)  (K=25, zero-padded) ----
    g_gemm(stream, eat_raw, P + P_MP_W, P + P_MP_B, eaT, NEDGE, 64, EDIM, 1);
    g_gemm(stream, ear_raw, P + P_MP_W, P + P_MP_B, eaR, NEDGE, 64, EDIM, 1);

    // ---- 3 TransformerConv layers ----
    for (int l = 0; l < NL; ++l) {
        const float* CT = P + P_CONV_T;
        const float* CR = P + P_CONV_R;
        size_t wl = (size_t)l * 64 * 64, bl = (size_t)l * 64;

        // ===== conv_t: src=guide, dst=site =====
        g_gemm(stream, eaT, CT + CV_We + wl, CT + CV_be + bl, eP,  NEDGE,   64, 64, 0);
        g_gemm(stream, hs,  CT + CV_Wq + wl, CT + CV_bq + bl, qS,  N_SITE,  64, 64, 0);
        g_gemm(stream, hs,  CT + CV_Ws + wl, CT + CV_bs + bl, skS, N_SITE,  64, 64, 0);
        g_gemm(stream, hg,  CT + CV_Wk + wl, CT + CV_bk + bl, kG,  N_GUIDE, 64, 64, 0);
        g_gemm(stream, hg,  CT + CV_Wv + wl, CT + CV_bv + bl, vG,  N_GUIDE, 64, 64, 0);
        g_fillu(stream, mS, 0u, (long)N_SITE * 4);
        g_fillf(stream, zS, 0.f, (long)N_SITE * 4);
        g_fillf(stream, aggS, 0.f, (long)N_SITE * 64);
        attn_pass1<<<e_blocks, 256, 0, stream>>>(qS, kG, eP, ei_src, ei_dst, scr, mS, NEDGE);
        attn_pass2<<<e_blocks, 256, 0, stream>>>(vG, eP, ei_src, ei_dst, scr, mS, zS, aggS, NEDGE);

        // ===== conv_r: src=site, dst=guide (uses pre-update hs/hg) =====
        g_gemm(stream, eaR, CR + CV_We + wl, CR + CV_be + bl, eP,  NEDGE,   64, 64, 0);
        g_gemm(stream, hg,  CR + CV_Wq + wl, CR + CV_bq + bl, qG,  N_GUIDE, 64, 64, 0);
        g_gemm(stream, hg,  CR + CV_Ws + wl, CR + CV_bs + bl, skG, N_GUIDE, 64, 64, 0);
        g_gemm(stream, hs,  CR + CV_Wk + wl, CR + CV_bk + bl, kS,  N_SITE,  64, 64, 0);
        g_gemm(stream, hs,  CR + CV_Wv + wl, CR + CV_bv + bl, vS,  N_SITE,  64, 64, 0);
        g_fillu(stream, mG, 0u, (long)N_GUIDE * 4);
        g_fillf(stream, zG, 0.f, (long)N_GUIDE * 4);
        g_fillf(stream, aggG, 0.f, (long)N_GUIDE * 64);
        attn_pass1<<<e_blocks, 256, 0, stream>>>(qG, kS, eP, ei_dst, ei_src, scr, mG, NEDGE);
        attn_pass2<<<e_blocks, 256, 0, stream>>>(vS, eP, ei_dst, ei_src, scr, mG, zG, aggG, NEDGE);

        // ===== residual + LN updates (after both directions read old h) =====
        node_update<<<ln_blocks_S, 256, 0, stream>>>(aggS, zS, skS,
            P + P_LNL_S_G + bl, P + P_LNL_S_B + bl, hs, N_SITE);
        node_update<<<ln_blocks_G, 256, 0, stream>>>(aggG, zG, skG,
            P + P_LNL_G_G + bl, P + P_LNL_G_B + bl, hg, N_GUIDE);
    }

    // ---- decoder ----
    // h1 = relu(bn1(ef @ W1 + b1)); h2 = relu(bn2(h1 @ W2 + b2))
    g_gemm(stream, efeat, P + P_EF_W1, P + P_EF_B1, h1, E_LAB, 64, EDIM, 2,
           P + P_EF_BN1_G, P + P_EF_BN1_B, P + P_EF_BN1_M, P + P_EF_BN1_V);
    g_gemm(stream, h1, P + P_EF_W2, P + P_EF_B2, h2, E_LAB, 64, 64, 2,
           P + P_EF_BN2_G, P + P_EF_BN2_B, P + P_EF_BN2_M, P + P_EF_BN2_V);
    // zcat = [hg[gi] | hs[si] | h2]
    gather_concat<<<(E_LAB + 7) / 8, 256, 0, stream>>>(hg, hs, h2, lgi, lsi, zcat, E_LAB);
    // d1 = LN(relu(zcat @ W1 + b1)); d2 = LN(relu(d1 @ W2 + b2))
    g_gemm(stream, zcat, P + P_DEC_W1, P + P_DEC_B1, dd1, E_LAB, 128, 192, 1);
    ln_kernel<128><<<(E_LAB + 7) / 8, 256, 0, stream>>>(dd1, dd1, P + P_DEC_LN1_G, P + P_DEC_LN1_B, E_LAB);
    g_gemm(stream, dd1, P + P_DEC_W2, P + P_DEC_B2, dd2, E_LAB, 64, 128, 1);
    ln_kernel<64><<<(E_LAB + 7) / 8, 256, 0, stream>>>(dd2, dd2, P + P_DEC_LN2_G, P + P_DEC_LN2_B, E_LAB);
    // out = d2 @ W3 + b3
    final_dot<<<(E_LAB + 7) / 8, 256, 0, stream>>>(dd2, P + P_DEC_W3, P + P_DEC_B3, out, E_LAB);
}
